// Kernelution_40999757808099
// MI455X (gfx1250) — matrix operand order repaired and checked by reading, not compiled, not run on hardware
//
#include <hip/hip_runtime.h>
#include <math.h>

#define B_   4
#define N_   8192
#define Dd_  448
#define BN_  (B_ * N_)
#define KT_  14       // 448 / 32 K-tiles
#define NT_  28       // 448 / 16 N-tiles
#define TN_  128      // n-tile for band kernel

typedef _Float16 v16h __attribute__((ext_vector_type(16)));
typedef _Float16 v8h  __attribute__((ext_vector_type(8)));
typedef float    v8f  __attribute__((ext_vector_type(8)));

#define SHUF16(lo, hi) __builtin_shufflevector((lo), (hi), 0,1,2,3,4,5,6,7,8,9,10,11,12,13,14,15)

// ---------------------------------------------------------------------------
// Weight repack: Wp[((kt*28+nt)*32+lane)*16 + e] = W[nt*16 + (lane&15)][kt*32 + (lane>>4)*16 + e]
// This is the per-lane 16x16x32 f16 WMMA B operand in contiguous k order, matching the A loads (B[k][n] = W[n][k]).
// Strides: lane -> 32 B, nt -> 1024 B, kt -> 28672 B (all immediate-offset friendly).
// ---------------------------------------------------------------------------
__global__ __launch_bounds__(256) void pack_w_kernel(const float* __restrict__ W,
                                                     _Float16* __restrict__ Wp) {
  int o = blockIdx.x * 256 + threadIdx.x;
  if (o >= KT_ * NT_ * 32 * 16) return;
  int e    = o & 15;
  int lane = (o >> 4) & 31;
  int nt   = (o >> 9) % NT_;
  int kt   = (o >> 9) / NT_;
  int n = nt * 16 + (lane & 15);
  int k = kt * 32 + ((lane >> 4) << 4) + e;
  Wp[o] = (_Float16)W[n * Dd_ + k];
}

// ---------------------------------------------------------------------------
// Fused GEMM tile: 16 rows x 448 cols = sin(A @ W^T + bias), A in LDS (f16),
// result written to LDS (f16). 4 waves, each owns 7 N-tiles; 14 K-steps of
// v_wmma_f32_16x16x32_f16 with f32 accumulation. Addressing is one base
// pointer per operand + constant offsets so loads fold immediates.
// ---------------------------------------------------------------------------
__device__ __forceinline__ void gemm_sin_stage(const _Float16* As,
                                               const _Float16* __restrict__ Wp,
                                               const float* __restrict__ bias,
                                               _Float16* Outs) {
  const int lane  = threadIdx.x & 31;
  const int wave  = threadIdx.x >> 5;
  const int mrow  = lane & 15;
  const int khalf = (lane >> 4) << 4;  // 0 or 16 (halves)

  v8f acc[7];
#pragma unroll
  for (int j = 0; j < 7; ++j) acc[j] = (v8f){0.f, 0.f, 0.f, 0.f, 0.f, 0.f, 0.f, 0.f};

  // Per-thread bases (element units of _Float16):
  //   A: row base + khalf; per-kt advance = 32 halves (64 B); halves [0..15].
  //   B: lane*16 + (wave*7)*512; per-j offset = 512 halves (1024 B); per-kt = 14336 halves.
  const _Float16* ap = As + mrow * Dd_ + khalf;
  const _Float16* bp = Wp + (size_t)lane * 16 + (size_t)(wave * 7) * 512;

  for (int kt = 0; kt < KT_; ++kt) {
    v8h alo = *(const v8h*)(ap);
    v8h ahi = *(const v8h*)(ap + 8);
    v16h a = SHUF16(alo, ahi);
#pragma unroll
    for (int j = 0; j < 7; ++j) {
      v8h blo = *(const v8h*)(bp + j * 512);
      v8h bhi = *(const v8h*)(bp + j * 512 + 8);
      v16h bb = SHUF16(blo, bhi);
      acc[j] = __builtin_amdgcn_wmma_f32_16x16x32_f16(false, a, false, bb,
                                                      (short)0, acc[j], false, false);
    }
    ap += 32;      // next 32-wide K slab in the LDS row
    bp += 14336;   // next kt block (28*512 halves)
  }

  // C/D layout: VGPR r, lanes 0-15 -> row r, lanes 16-31 -> row r+8; col = nt*16 + lane&15
  const int rbase = (lane >> 4) << 3;
  const int cl = lane & 15;
  _Float16* obase = Outs + rbase * Dd_;
#pragma unroll
  for (int j = 0; j < 7; ++j) {
    const int col = (wave * 7 + j) * 16 + cl;
    const float bv = bias[col];
#pragma unroll
    for (int r = 0; r < 8; ++r) {
      float v = __builtin_sinf(acc[j][r] + bv);
      obase[r * Dd_ + col] = (_Float16)v;
    }
  }
}

// ---------------------------------------------------------------------------
// Fused two-layer MLP: sin(GEMM1)->scalenorm->sin(GEMM2)-> eig (mode 0) or
// fixnorm + 7-way pooled params (mode 1). One block = 16 rows, 128 threads.
// ---------------------------------------------------------------------------
__global__ __launch_bounds__(128) void mlp_kernel(const float* __restrict__ X,
                                                  const _Float16* __restrict__ Wp1,
                                                  const float* __restrict__ b1,
                                                  const _Float16* __restrict__ Wp2,
                                                  const float* __restrict__ b2,
                                                  const float* __restrict__ gptr,
                                                  int mode,
                                                  float* __restrict__ outEig,
                                                  float* __restrict__ outPooled) {
  __shared__ _Float16 Arow[16 * Dd_];
  __shared__ _Float16 Brow[16 * Dd_];
  __shared__ float rowpart[16][8];
  __shared__ float rowscale[16];

  const int tid = threadIdx.x;
  const size_t row0 = (size_t)blockIdx.x * 16;

  // load 16 rows of X, convert f32 -> f16 into LDS
  for (int i = tid; i < 16 * Dd_; i += 128) Arow[i] = (_Float16)X[row0 * Dd_ + i];
  __syncthreads();

  gemm_sin_stage(Arow, Wp1, b1, Brow);
  __syncthreads();

  // scalenorm: g * x / max(||x||, eps), deterministic LDS tree reduction
  const int row = tid >> 3, sub = tid & 7;
  {
    float ss = 0.f;
    const _Float16* p = &Brow[row * Dd_ + sub * 56];
    for (int i = 0; i < 56; ++i) { float v = (float)p[i]; ss += v * v; }
    rowpart[row][sub] = ss;
  }
  __syncthreads();
  if (tid < 16) {
    float s = 0.f;
#pragma unroll
    for (int j = 0; j < 8; ++j) s += rowpart[tid][j];
    rowscale[tid] = gptr[0] / fmaxf(sqrtf(s), 1e-5f);
  }
  __syncthreads();
  for (int i = tid; i < 16 * Dd_; i += 128)
    Brow[i] = (_Float16)((float)Brow[i] * rowscale[i / Dd_]);
  __syncthreads();

  gemm_sin_stage(Brow, Wp2, b2, Arow);
  __syncthreads();

  if (mode == 0) {
    // eig = mean over D
    float s = 0.f;
    const _Float16* p = &Arow[row * Dd_ + sub * 56];
    for (int i = 0; i < 56; ++i) s += (float)p[i];
    rowpart[row][sub] = s;
    __syncthreads();
    if (tid < 16) {
      float s2 = 0.f;
#pragma unroll
      for (int j = 0; j < 8; ++j) s2 += rowpart[tid][j];
      outEig[row0 + tid] = s2 * (1.0f / 448.0f);
    }
  } else {
    // fixnorm + AdaptiveAvgPool1d(7) (448/7 = 64 per channel)
    float ss = 0.f;
    const _Float16* p = &Arow[row * Dd_ + sub * 56];
    for (int i = 0; i < 56; ++i) { float v = (float)p[i]; ss += v * v; }
    rowpart[row][sub] = ss;
    __syncthreads();
    if (tid < 16) {
      float s2 = 0.f;
#pragma unroll
      for (int j = 0; j < 8; ++j) s2 += rowpart[tid][j];
      rowscale[tid] = 1.0f / fmaxf(sqrtf(s2), 1e-5f);
    }
    __syncthreads();
    if (tid < 112) {
      int r = tid / 7, c = tid % 7;
      float s2 = 0.f;
      const _Float16* q = &Arow[r * Dd_ + c * 64];
      for (int i = 0; i < 64; ++i) s2 += (float)q[i];
      outPooled[(row0 + r) * 7 + c] = s2 * rowscale[r] * (1.0f / 64.0f);
    }
  }
}

// ---------------------------------------------------------------------------
// Complex helpers + Givens / banded-coefficient precomputation.
// ---------------------------------------------------------------------------
__device__ __forceinline__ float2 cmul(float2 a, float2 b) {
  return make_float2(a.x * b.x - a.y * b.y, a.x * b.y + a.y * b.x);
}
__device__ __forceinline__ float2 cneg(float2 a) { return make_float2(-a.x, -a.y); }
__device__ __forceinline__ float2 cmad(float2 c, float2 x, float2 acc) {
  acc.x += c.x * x.x - c.y * x.y;
  acc.y += c.x * x.y + c.y * x.x;
  return acc;
}

struct Giv { float2 ii, ij, ji, jj; };

__device__ __forceinline__ Giv givensf(float al, float be, float ga) {
  const float PIF = 3.14159265358979323846f;
  float A = (al + be) * PIF, Bb = (al - be) * PIF, C = ga * PIF;
  float sA = __builtin_sinf(A),  cA = __builtin_cosf(A);
  float sB = __builtin_sinf(Bb), cB = __builtin_cosf(Bb);
  float sC = __builtin_sinf(C),  cC = __builtin_cosf(C);
  Giv g;
  g.ii = make_float2(cA * cC, -sA * cC);   //  exp(-iA)  cosC
  g.ij = make_float2(-cB * sC, -sB * sC);  // -exp(+iBb) sinC
  g.ji = make_float2(cB * sC, -sB * sC);   //  exp(-iBb) sinC
  g.jj = make_float2(cA * cC, sA * cC);    //  exp(+iA)  cosC
  return g;
}

enum { cUA = 0, cUB, cUC, cLA, cLB, cLC, cIUA, cIUB, cIUC, cILA, cILB, cILC, cDG, cIDG, C_NUM };

__global__ __launch_bounds__(256) void coef_kernel(const float* __restrict__ pooled,
                                                   float2* __restrict__ coef) {
  int t = blockIdx.x * 256 + threadIdx.x;
  if (t >= BN_) return;
  int b = t / N_, n = t % N_;
  const float* pf = pooled + (size_t)b * N_ * 7;
#define CH(j, i) pf[(j) * N_ + (i)]
  const float2 z0 = make_float2(0.f, 0.f);
  Giv gun{}, gup{}, gln{}, glp{};
  bool hasN = (n <= N_ - 2), hasP = (n >= 1);
  if (hasN) {
    gun = givensf(CH(3, n + 1), CH(4, n + 1), CH(5, n + 1));  // Gu[n]   = G(ch[3..5][n+1])
    gln = givensf(CH(0, n),     CH(1, n),     CH(2, n));      // Gl[n]   = G(ch[0..2][n])
  }
  if (hasP) {
    gup = givensf(CH(3, n),     CH(4, n),     CH(5, n));      // Gu[n-1] = G(ch[3..5][n])
    glp = givensf(CH(0, n - 1), CH(1, n - 1), CH(2, n - 1));  // Gl[n-1]
  }
  float2 uA, uB, uC, lA, lB, lC, iuA, iuB, iuC, ilA, ilB, ilC;
  if (n == 0) {
    uA = z0;        uB = gun.ii;               uC = gun.ij;
    lA = z0;        lB = gln.ii;               lC = gln.ij;
    iuA = z0;       iuB = gln.jj;              iuC = cneg(gln.ij);
    ilA = z0;       ilB = gun.jj;              ilC = cneg(gun.ij);
  } else if (n == N_ - 1) {
    uA = gup.ji;        uB = gup.jj;           uC = z0;
    lA = glp.ji;        lB = glp.jj;           lC = z0;
    iuA = cneg(glp.ji); iuB = glp.ii;          iuC = z0;
    ilA = cneg(gup.ji); ilB = gup.ii;          ilC = z0;
  } else {
    uA = gup.ji;
    uB = cmul(gup.jj, gun.ii);
    uC = cmul(gup.jj, gun.ij);
    lA = cmul(gln.ii, glp.ji);
    lB = cmul(gln.ii, glp.jj);
    lC = gln.ij;
    iuA = cneg(glp.ji);                        // G'u = conj_trs(Gl)
    iuB = cmul(glp.ii, gln.jj);
    iuC = cneg(cmul(glp.ii, gln.ij));
    ilA = cneg(cmul(gun.jj, gup.ji));          // G'l = conj_trs(Gu)
    ilB = cmul(gun.jj, gup.ii);
    ilC = cneg(gun.ij);
  }
  float th = CH(6, n) * 3.14159265358979323846f;
  float2 dg = make_float2(__builtin_cosf(th), __builtin_sinf(th));
  coef[(size_t)cUA * BN_ + t] = uA;   coef[(size_t)cUB * BN_ + t] = uB;
  coef[(size_t)cUC * BN_ + t] = uC;   coef[(size_t)cLA * BN_ + t] = lA;
  coef[(size_t)cLB * BN_ + t] = lB;   coef[(size_t)cLC * BN_ + t] = lC;
  coef[(size_t)cIUA * BN_ + t] = iuA; coef[(size_t)cIUB * BN_ + t] = iuB;
  coef[(size_t)cIUC * BN_ + t] = iuC; coef[(size_t)cILA * BN_ + t] = ilA;
  coef[(size_t)cILB * BN_ + t] = ilB; coef[(size_t)cILC * BN_ + t] = ilC;
  coef[(size_t)cDG * BN_ + t] = dg;
  coef[(size_t)cIDG * BN_ + t] = make_float2(dg.x, -dg.y);
#undef CH
}

// ---------------------------------------------------------------------------
// Chebyshev filter (K=8) with Gibbs damping.
// ---------------------------------------------------------------------------
__global__ __launch_bounds__(256) void cheb_kernel(const float* __restrict__ eig,
                                                   const float* __restrict__ cheb,
                                                   const float* __restrict__ gd,
                                                   float* __restrict__ filt) {
  int t = blockIdx.x * 256 + threadIdx.x;
  if (t >= BN_) return;
  int b = t / N_;
  const float* c = cheb + b * 9;
  const float* g = gd + b * 9;
  float x = eig[t];
  float T0 = 1.f, T1 = x;
  float f = c[0] + T1 * c[1] * g[1];
#pragma unroll
  for (int k = 2; k <= 8; ++k) {
    float T2 = 2.f * x * T1 - T0;
    f += T2 * c[k] * g[k];
    T0 = T1; T1 = T2;
  }
  filt[t] = f;
}

// ---------------------------------------------------------------------------
// Banded DHHP sweep (U then L), fused with permute/Diag/filt.
// forward (inverse==0): gather-permute real X -> complex, U, L, *Diag*filt -> Zout
// inverse (inverse==1): Zin * conj(Diag), U (conj_trs Gl), L (conj_trs Gu),
//                       inverse-permute scatter of real part -> outReal
// One thread per d-column, TN_=128 n's per block with 2-halo streaming.
// ---------------------------------------------------------------------------
__global__ __launch_bounds__(Dd_) void band_kernel(const float* __restrict__ Xreal,
                                                   const float2* __restrict__ Zin,
                                                   float2* __restrict__ Zout,
                                                   float* __restrict__ outReal,
                                                   const float2* __restrict__ uA,
                                                   const float2* __restrict__ uB,
                                                   const float2* __restrict__ uC,
                                                   const float2* __restrict__ lA,
                                                   const float2* __restrict__ lB,
                                                   const float2* __restrict__ lC,
                                                   const float2* __restrict__ Dg,
                                                   const float* __restrict__ filt,
                                                   int inverse) {
  __shared__ float2 sUA[TN_ + 2], sUB[TN_ + 2], sUC[TN_ + 2];
  __shared__ float2 sLA[TN_], sLB[TN_], sLC[TN_], sDg[TN_];
  __shared__ float sF[TN_];

  const int d = threadIdx.x;
  const int b = blockIdx.y;
  const int n0 = blockIdx.x * TN_;
  const size_t base = (size_t)b * N_;
  const size_t rowbase = base * Dd_;

  for (int i = threadIdx.x; i < TN_ + 2; i += Dd_) {
    int k = n0 - 1 + i;
    if (k >= 0 && k < N_) {
      sUA[i] = uA[base + k]; sUB[i] = uB[base + k]; sUC[i] = uC[base + k];
    } else {
      float2 z = make_float2(0.f, 0.f);
      sUA[i] = z; sUB[i] = z; sUC[i] = z;
    }
  }
  for (int i = threadIdx.x; i < TN_; i += Dd_) {
    int n = n0 + i;
    sLA[i] = lA[base + n]; sLB[i] = lB[base + n]; sLC[i] = lC[base + n];
    sDg[i] = Dg[base + n];
    sF[i] = filt ? filt[base + n] : 1.0f;
  }
  __syncthreads();

  auto loadX = [&](int k) -> float2 {
    float2 r = make_float2(0.f, 0.f);
    if (k < 0 || k >= N_) return r;
    if (!inverse) {
      int src = ((k & 127) << 6) + (k >> 7);  // forward permute gather
      r.x = Xreal[rowbase + (size_t)src * Dd_ + d];
    } else {
      float2 v = Zin[rowbase + (size_t)k * Dd_ + d];
      r = cmul(Dg[base + k], v);              // pre-multiply by conj(Diag)
    }
    return r;
  };
  auto Yat = [&](int k, float2 xm, float2 xc, float2 xp) -> float2 {
    int i = k - (n0 - 1);
    float2 y = make_float2(0.f, 0.f);
    y = cmad(sUA[i], xm, y);
    y = cmad(sUB[i], xc, y);
    y = cmad(sUC[i], xp, y);
    return y;
  };

  const int kstart = n0 - 1;
  float2 xm = loadX(kstart - 1);
  float2 xc = loadX(kstart);
  float2 xp = loadX(kstart + 1);
  float2 Ym = (kstart >= 0) ? Yat(kstart, xm, xc, xp) : make_float2(0.f, 0.f);
  xm = xc; xc = xp; xp = loadX(n0 + 1);
  float2 Yc = Yat(n0, xm, xc, xp);

  for (int n = n0; n < n0 + TN_; ++n) {
    int k = n + 1;
    xm = xc; xc = xp; xp = loadX(k + 1);
    float2 Yp = (k < N_) ? Yat(k, xm, xc, xp) : make_float2(0.f, 0.f);
    int li = n - n0;
    float2 z = make_float2(0.f, 0.f);
    z = cmad(sLA[li], Ym, z);
    z = cmad(sLB[li], Yc, z);
    z = cmad(sLC[li], Yp, z);
    if (!inverse) {
      z = cmul(sDg[li], z);                   // transform=True: Diag after sweeps
      float f = sF[li];
      z.x *= f; z.y *= f;                     // Chebyshev filter scaling
      Zout[rowbase + (size_t)n * Dd_ + d] = z;
    } else {
      int dest = ((n & 127) << 6) + (n >> 7); // inverse permute scatter
      outReal[rowbase + (size_t)dest * Dd_ + d] = z.x;
    }
    Ym = Yc; Yc = Yp;
  }
}

// ---------------------------------------------------------------------------
// Host launcher.
// ---------------------------------------------------------------------------
extern "C" void kernel_launch(void* const* d_in, const int* in_sizes, int n_in,
                              void* d_out, int out_size, void* d_ws, size_t ws_size,
                              hipStream_t stream) {
  (void)in_sizes; (void)n_in; (void)out_size; (void)ws_size;
  const float* X    = (const float*)d_in[0];
  const float* W1e  = (const float*)d_in[1];
  const float* b1e  = (const float*)d_in[2];
  const float* ge   = (const float*)d_in[3];
  const float* W2e  = (const float*)d_in[4];
  const float* b2e  = (const float*)d_in[5];
  const float* W1p  = (const float*)d_in[6];
  const float* b1p  = (const float*)d_in[7];
  const float* gp   = (const float*)d_in[8];
  const float* W2p  = (const float*)d_in[9];
  const float* b2p  = (const float*)d_in[10];
  const float* cheb = (const float*)d_in[11];
  const float* gd   = (const float*)d_in[12];
  float* out = (float*)d_out;

  char* ws = (char*)d_ws;
  size_t off = 0;
  auto alloc = [&](size_t bytes) -> char* {
    char* p = ws + off;
    off += (bytes + 255) & ~(size_t)255;
    return p;
  };
  const size_t wpBytes = (size_t)KT_ * NT_ * 32 * 16 * sizeof(_Float16);  // 448*448 f16
  _Float16* wp1e = (_Float16*)alloc(wpBytes);
  _Float16* wp2e = (_Float16*)alloc(wpBytes);
  _Float16* wp1p = (_Float16*)alloc(wpBytes);
  _Float16* wp2p = (_Float16*)alloc(wpBytes);
  float* eig    = (float*)alloc((size_t)BN_ * sizeof(float));
  float* filt   = (float*)alloc((size_t)BN_ * sizeof(float));
  float* pooled = (float*)alloc((size_t)BN_ * 7 * sizeof(float));
  float2* coef  = (float2*)alloc((size_t)C_NUM * BN_ * sizeof(float2));
  float2* Zc    = (float2*)alloc((size_t)BN_ * Dd_ * sizeof(float2));

  const int packBlocks = (KT_ * NT_ * 32 * 16 + 255) / 256;
  pack_w_kernel<<<packBlocks, 256, 0, stream>>>(W1e, wp1e);
  pack_w_kernel<<<packBlocks, 256, 0, stream>>>(W2e, wp2e);
  pack_w_kernel<<<packBlocks, 256, 0, stream>>>(W1p, wp1p);
  pack_w_kernel<<<packBlocks, 256, 0, stream>>>(W2p, wp2p);

  const int mlpBlocks = BN_ / 16;  // 2048
  mlp_kernel<<<mlpBlocks, 128, 0, stream>>>(X, wp1e, b1e, wp2e, b2e, ge, 0, eig, nullptr);
  mlp_kernel<<<mlpBlocks, 128, 0, stream>>>(X, wp1p, b1p, wp2p, b2p, gp, 1, nullptr, pooled);

  const int elBlocks = (BN_ + 255) / 256;
  cheb_kernel<<<elBlocks, 256, 0, stream>>>(eig, cheb, gd, filt);
  coef_kernel<<<elBlocks, 256, 0, stream>>>(pooled, coef);

  auto C = [&](int j) { return coef + (size_t)j * BN_; };
  dim3 bandGrid(N_ / TN_, B_);
  // forward: permute-gather X, U(Gu), L(Gl), *Diag, *filt -> Zc
  band_kernel<<<bandGrid, Dd_, 0, stream>>>(X, nullptr, Zc, nullptr,
                                            C(cUA), C(cUB), C(cUC),
                                            C(cLA), C(cLB), C(cLC),
                                            C(cDG), filt, 0);
  // inverse: Zc * conj(Diag), U(conj_trs Gl), L(conj_trs Gu), inv-permute, real -> out
  band_kernel<<<bandGrid, Dd_, 0, stream>>>(nullptr, Zc, nullptr, out,
                                            C(cIUA), C(cIUB), C(cIUC),
                                            C(cILA), C(cILB), C(cILC),
                                            C(cIDG), nullptr, 1);
}
